// EfficientCorrBlock_69853348102330
// MI455X (gfx1250) — matrix operand order repaired and checked by reading, not compiled, not run on hardware
//
#include <hip/hip_runtime.h>

// ---------------------------------------------------------------------------
// EfficientCorrBlock for MI455X (gfx1250, wave32, WMMA + async LDS copies)
//
// Pipeline:
//   1) feat1_kernel    : bilinear sample fmap1 @ coords1, L2-normalize -> f16
//   2) down_kernel x3  : cascaded [1,3,3,1]/8 antialiased 2x downsample (fp32)
//   3) normpix_kernel  : per-pixel L2-normalize each pyramid level -> f16
//   4) corr_kernel     : per 16-query tile, double-buffered
//                        GLOBAL_LOAD_ASYNC_TO_LDS_B128 staging feeding
//                        v_wmma_f32_16x16x32_f16 K-chains that produce the
//                        8x8 corner-dot grid per (query, level); VALU phase
//                        applies clipped bilinear weights -> (B,N,196) fp32
// ---------------------------------------------------------------------------

typedef __attribute__((ext_vector_type(16))) _Float16 v16h;
typedef __attribute__((ext_vector_type(8)))  float    v8f;

#define BATCH 2
#define HH0   64
#define WW0   64
#define CH    256
#define NQ    4096
#define RAD   3
#define KOFF  49   // (2*RAD+1)^2
#define NLVL  4

union FragH { v16h v; unsigned int u[8]; _Float16 h[16]; };

__device__ __forceinline__ float wave_reduce_add(float v) {
  #pragma unroll
  for (int off = 16; off > 0; off >>= 1) v += __shfl_xor(v, off, 32);
  return v;
}

__device__ __forceinline__ int iclamp(int v, int lo, int hi) {
  return min(max(v, lo), hi);
}

// Async global->LDS 16B copy (ASYNCcnt-tracked; no VGPR round trip).
__device__ __forceinline__ void async_copy_b128(unsigned lds_addr,
                                                unsigned long long gaddr) {
  asm volatile("global_load_async_to_lds_b128 %0, %1, off"
               :: "v"(lds_addr), "v"(gaddr));
}

// -------------------- 1) feat1: bilinear sample + normalize ------------------
__global__ __launch_bounds__(256) void feat1_kernel(
    const float* __restrict__ fmap, const float* __restrict__ coords,
    _Float16* __restrict__ outh) {
  const int wave = threadIdx.x >> 5, lane = threadIdx.x & 31;
  const int q = blockIdx.x * 8 + wave;
  const int b = blockIdx.y;

  const float2 cc = ((const float2*)coords)[(size_t)b * NQ + q];
  const float x = cc.x, y = cc.y;
  const int xf = (int)floorf(x), yf = (int)floorf(y);
  const int x0 = iclamp(xf, 0, WW0 - 1), x1 = iclamp(xf + 1, 0, WW0 - 1);
  const int y0 = iclamp(yf, 0, HH0 - 1), y1 = iclamp(yf + 1, 0, HH0 - 1);
  const float x0f = (float)x0, x1f = (float)x1, y0f = (float)y0, y1f = (float)y1;
  const float wa = (x1f - x) * (y1f - y), wb = (x1f - x) * (y - y0f);
  const float wc = (x - x0f) * (y1f - y), wd = (x - x0f) * (y - y0f);

  const float* basep = fmap + (size_t)b * HH0 * WW0 * CH;
  const int ch = lane * 8;
  const float* pa = basep + ((size_t)y0 * WW0 + x0) * CH + ch;
  const float* pb = basep + ((size_t)y1 * WW0 + x0) * CH + ch;
  const float* pc = basep + ((size_t)y0 * WW0 + x1) * CH + ch;
  const float* pd = basep + ((size_t)y1 * WW0 + x1) * CH + ch;

  float A[8], B[8], C[8], D[8], v[8];
  *(float4*)&A[0] = *(const float4*)(pa); *(float4*)&A[4] = *(const float4*)(pa + 4);
  *(float4*)&B[0] = *(const float4*)(pb); *(float4*)&B[4] = *(const float4*)(pb + 4);
  *(float4*)&C[0] = *(const float4*)(pc); *(float4*)&C[4] = *(const float4*)(pc + 4);
  *(float4*)&D[0] = *(const float4*)(pd); *(float4*)&D[4] = *(const float4*)(pd + 4);

  float ss = 0.f;
  #pragma unroll
  for (int j = 0; j < 8; ++j) {
    v[j] = wa * A[j] + wb * B[j] + wc * C[j] + wd * D[j];
    ss += v[j] * v[j];
  }
  ss = wave_reduce_add(ss);
  const float inv = 1.0f / (sqrtf(ss) + 1e-6f);

  _Float16 h[8];
  #pragma unroll
  for (int j = 0; j < 8; ++j) h[j] = (_Float16)(v[j] * inv);
  *(uint4*)(outh + ((size_t)b * NQ + q) * CH + ch) = *(uint4*)h;
}

// -------------------- 2) antialiased 2x bilinear downsample ------------------
__global__ __launch_bounds__(256) void down_kernel(
    const float* __restrict__ in, float* __restrict__ out, int Hi, int Wi) {
  const int c = threadIdx.x;
  const int Ho = Hi >> 1, Wo = Wi >> 1;
  const int pix = blockIdx.x, b = blockIdx.y;
  const int ho = pix / Wo, wo = pix - ho * Wo;

  const float wt[4] = {0.125f, 0.375f, 0.375f, 0.125f};
  float wy[4], wx[4]; int iy[4], ix[4];
  float sy = 0.f, sx = 0.f;
  #pragma unroll
  for (int t = 0; t < 4; ++t) {
    int yy = 2 * ho - 1 + t;
    bool vy = (yy >= 0) && (yy < Hi);
    iy[t] = iclamp(yy, 0, Hi - 1);
    wy[t] = vy ? wt[t] : 0.f; sy += wy[t];
    int xx = 2 * wo - 1 + t;
    bool vx = (xx >= 0) && (xx < Wi);
    ix[t] = iclamp(xx, 0, Wi - 1);
    wx[t] = vx ? wt[t] : 0.f; sx += wx[t];
  }
  const float* basep = in + (size_t)b * Hi * Wi * CH + c;
  float acc = 0.f;
  #pragma unroll
  for (int ty = 0; ty < 4; ++ty) {
    #pragma unroll
    for (int tx = 0; tx < 4; ++tx) {
      acc += wy[ty] * wx[tx] * basep[((size_t)iy[ty] * Wi + ix[tx]) * CH];
    }
  }
  out[(((size_t)b * Ho + ho) * Wo + wo) * CH + c] = acc / (sy * sx);
}

// -------------------- 3) per-pixel normalize -> f16 --------------------------
__global__ __launch_bounds__(256) void normpix_kernel(
    const float* __restrict__ in, _Float16* __restrict__ outh, int P) {
  const int wave = threadIdx.x >> 5, lane = threadIdx.x & 31;
  const int pix = blockIdx.x * 8 + wave, b = blockIdx.y;
  if (pix >= P) return;
  const float* row = in + ((size_t)b * P + pix) * CH;
  const int ch = lane * 8;
  float v[8];
  *(float4*)&v[0] = *(const float4*)(row + ch);
  *(float4*)&v[4] = *(const float4*)(row + ch + 4);
  float ss = 0.f;
  #pragma unroll
  for (int j = 0; j < 8; ++j) ss += v[j] * v[j];
  ss = wave_reduce_add(ss);
  const float inv = 1.0f / (sqrtf(ss) + 1e-6f);
  _Float16 h[8];
  #pragma unroll
  for (int j = 0; j < 8; ++j) h[j] = (_Float16)(v[j] * inv);
  *(uint4*)(outh + ((size_t)b * P + pix) * CH + ch) = *(uint4*)h;
}

// -------------------- 4) WMMA correlation kernel -----------------------------
// Block = 128 threads (4 waves). Each block: 16 queries of one batch.
__global__ __launch_bounds__(128) void corr_kernel(
    const float* __restrict__ coords2, const _Float16* __restrict__ feat1h,
    const _Float16* __restrict__ f2n0, const _Float16* __restrict__ f2n1,
    const _Float16* __restrict__ f2n2, const _Float16* __restrict__ f2n3,
    float* __restrict__ out) {
  __shared__ _Float16 ldsA[16 * CH];            // feat1 tile: 16x256 f16 (8 KB)
  __shared__ _Float16 ldsB[4][2][16 * CH];      // per-wave double buffer (64 KB)
  __shared__ float    ldsDot[16][68];           // 8x8 corner-dot grid per query
  __shared__ float    ldsQx[16], ldsQy[16];
  __shared__ int      ldsQfx[16], ldsQfy[16];

  const int tid  = threadIdx.x;
  const int wave = tid >> 5, lane = tid & 31;
  const int qbase = blockIdx.x * 16, b = blockIdx.y;

  // Stage feat1 tile (contiguous 8 KB) into LDS: 512 x uint4 over 128 threads.
  {
    const uint4* src = (const uint4*)(feat1h + ((size_t)b * NQ + qbase) * CH);
    uint4* dst = (uint4*)ldsA;
    #pragma unroll
    for (int i = 0; i < 4; ++i) dst[i * 128 + tid] = src[i * 128 + tid];
  }
  __syncthreads();

  // Preload A fragments: 8 K-steps of V_WMMA_F32_16X16X32_F16.
  // A layout (16-bit A 16x32): lanes 0-15 row M=lane K{0..7,16..23},
  // lanes 16-31 row M=lane-16 K{8..15,24..31}; pairs per VGPR.
  FragH af[8];
  {
    const int r = lane & 15, g = lane >> 4;
    #pragma unroll
    for (int s = 0; s < 8; ++s) {
      #pragma unroll
      for (int v = 0; v < 8; ++v) {
        const int kb = ((v < 4) ? 0 : 16) + 8 * g + 2 * (v & 3);
        af[s].u[v] = *(const unsigned int*)&ldsA[r * CH + s * 32 + kb];
      }
    }
  }

  const _Float16* f2s[4] = {f2n0, f2n1, f2n2, f2n3};

  #pragma unroll
  for (int level = 0; level < NLVL; ++level) {
    const int Wl = WW0 >> level, Hl = HH0 >> level;
    const float scale = (float)(1 << level);
    const _Float16* f2 = f2s[level];

    if (tid < 16) {
      const float2 cc = ((const float2*)coords2)[(size_t)b * NQ + qbase + tid];
      const float cx = cc.x / scale, cy = cc.y / scale;
      ldsQx[tid] = cx; ldsQy[tid] = cy;
      ldsQfx[tid] = (int)floorf(cx); ldsQfy[tid] = (int)floorf(cy);
    }
    __syncthreads();

    // Issue the 16 async row copies (16 x 512 B) for chain `q` into buffer.
    auto issue_chain = [&](int q, int buf) {
      const int fbx = ldsQfx[q], fby = ldsQfy[q];
      const unsigned lbase =
          (unsigned)(size_t)(&ldsB[wave][buf][0]) + (unsigned)lane * 16u;
      #pragma unroll
      for (int j = 0; j < 16; ++j) {
        const int wy = 2 * wave + (j >> 3), wx = j & 7;  // tile t == wave
        const int px = iclamp(fbx - 3 + wx, 0, Wl - 1);
        const int py = iclamp(fby - 3 + wy, 0, Hl - 1);
        const unsigned long long ga =
            (unsigned long long)(size_t)(f2 +
                (((size_t)b * (Hl * Wl) + py * Wl + px) << 8)) +
            (unsigned long long)lane * 16ull;
        async_copy_b128(lbase + (unsigned)(j << 9), ga);
      }
    };

    // Software pipeline: 64 chains total, wave w owns window-row-pair t=w and
    // iterates q=0..15 with double-buffered async staging.
    issue_chain(0, 0);
    for (int i = 0; i < 16; ++i) {
      const int q = i, cur = i & 1;
      if (i < 15) {
        issue_chain(i + 1, cur ^ 1);
        // 32 async copies outstanding; in-order completion => <=16 left means
        // chain i's 16 copies have landed in LDS.
        asm volatile("s_wait_asynccnt 16" ::: "memory");
      } else {
        asm volatile("s_wait_asynccnt 0" ::: "memory");
      }

      // B layout (16-bit B 32x16): lanes 0-15 col=lane K{0..7,16..23},
      // lanes 16-31 col=lane-16 K{8..15,24..31}, consecutive K pairs per VGPR.
      const _Float16* bw = &ldsB[wave][cur][0];
      v8f c = {};
      const int col = lane & 15, bg = lane >> 4;
      #pragma unroll
      for (int s = 0; s < 8; ++s) {
        FragH bf;
        #pragma unroll
        for (int v = 0; v < 8; ++v)
          bf.u[v] = *(const unsigned int*)&bw[(col << 8) + (s << 5) + ((v < 4) ? 0 : 16) + (bg << 3) + ((v & 3) << 1)];
        c = __builtin_amdgcn_wmma_f32_16x16x32_f16(
            false, af[s].v, false, bf.v, (short)0, c, false, false);
      }

      // Extract D row q: D[m][n] -> VGPR (m&7), lane (n + 16*(m>=8 ? 1 : 0)).
      const int jj = lane & 15, dg = lane >> 4;
      #pragma unroll
      for (int v = 0; v < 8; ++v) {
        if (q == v + (dg << 3)) ldsDot[q][(wave << 4) + jj] = c[v];
      }
    }
    __syncthreads();

    // Combine: 16 queries x 49 offsets; clipped-corner bilinear on dot grid.
    const float Wm1 = (float)(Wl - 1), Hm1 = (float)(Hl - 1);
    #pragma unroll
    for (int i = 0; i < 7; ++i) {
      const int idx = i * 128 + tid;
      if (idx < 16 * KOFF) {
        const int q = idx / KOFF, kk = idx - q * KOFF;
        const int ky = kk / 7, kx = kk - ky * 7;
        const float dx = (float)(kx - RAD), dy = (float)(ky - RAD);
        const float cx = ldsQx[q], cy = ldsQy[q];
        const int fbx = ldsQfx[q], fby = ldsQfy[q];
        const float cxo = fminf(fmaxf(cx + dx, 0.f), Wm1);
        const float cyo = fminf(fmaxf(cy + dy, 0.f), Hm1);
        const int x0 = (int)cxo, y0 = (int)cyo;       // cxo,cyo >= 0
        const int x1 = min(x0 + 1, Wl - 1), y1 = min(y0 + 1, Hl - 1);
        const float wax = (float)x1 - cxo, wbx = cxo - (float)x0;
        const float way = (float)y1 - cyo, wby = cyo - (float)y0;
        const int sx0 = iclamp(x0 - fbx + 3, 0, 7), sx1 = iclamp(x1 - fbx + 3, 0, 7);
        const int sy0 = iclamp(y0 - fby + 3, 0, 7), sy1 = iclamp(y1 - fby + 3, 0, 7);
        const float val = wax * way * ldsDot[q][sy0 * 8 + sx0]
                        + wax * wby * ldsDot[q][sy1 * 8 + sx0]
                        + wbx * way * ldsDot[q][sy0 * 8 + sx1]
                        + wbx * wby * ldsDot[q][sy1 * 8 + sx1];
        out[((size_t)(b * NQ + qbase + q)) * (NLVL * KOFF) + level * KOFF + kk] = val;
      }
    }
    __syncthreads();
  }
}

// ---------------------------------------------------------------------------
extern "C" void kernel_launch(void* const* d_in, const int* in_sizes, int n_in,
                              void* d_out, int out_size, void* d_ws, size_t ws_size,
                              hipStream_t stream) {
  const float* fmap1   = (const float*)d_in[0];
  const float* fmap2   = (const float*)d_in[1];
  const float* coords1 = (const float*)d_in[2];
  const float* coords2 = (const float*)d_in[3];
  float* out = (float*)d_out;

  char* ws = (char*)d_ws;
  size_t off = 0;
  auto carve = [&](size_t bytes) -> char* {
    char* p = ws + off;
    off += (bytes + 255) & ~(size_t)255;
    return p;
  };

  _Float16* feat1h = (_Float16*)carve((size_t)BATCH * NQ * CH * 2);        // 4 MB
  float*    raw1   = (float*)  carve((size_t)BATCH * 32 * 32 * CH * 4);    // 2 MB
  float*    raw2   = (float*)  carve((size_t)BATCH * 16 * 16 * CH * 4);    // 0.5 MB
  float*    raw3   = (float*)  carve((size_t)BATCH *  8 *  8 * CH * 4);    // 128 KB
  _Float16* f2n0   = (_Float16*)carve((size_t)BATCH * 64 * 64 * CH * 2);   // 4 MB
  _Float16* f2n1   = (_Float16*)carve((size_t)BATCH * 32 * 32 * CH * 2);   // 1 MB
  _Float16* f2n2   = (_Float16*)carve((size_t)BATCH * 16 * 16 * CH * 2);   // 256 KB
  _Float16* f2n3   = (_Float16*)carve((size_t)BATCH *  8 *  8 * CH * 2);   // 64 KB
  (void)ws_size; (void)in_sizes; (void)n_in; (void)out_size;

  // 1) feat1
  feat1_kernel<<<dim3(NQ / 8, BATCH), 256, 0, stream>>>(fmap1, coords1, feat1h);

  // 2) cascaded pyramid (raw features, fp32)
  down_kernel<<<dim3(32 * 32, BATCH), 256, 0, stream>>>(fmap2, raw1, 64, 64);
  down_kernel<<<dim3(16 * 16, BATCH), 256, 0, stream>>>(raw1, raw2, 32, 32);
  down_kernel<<<dim3( 8 *  8, BATCH), 256, 0, stream>>>(raw2, raw3, 16, 16);

  // 3) per-pixel normalize -> f16
  normpix_kernel<<<dim3(4096 / 8, BATCH), 256, 0, stream>>>(fmap2, f2n0, 4096);
  normpix_kernel<<<dim3(1024 / 8, BATCH), 256, 0, stream>>>(raw1,  f2n1, 1024);
  normpix_kernel<<<dim3( 256 / 8, BATCH), 256, 0, stream>>>(raw2,  f2n2,  256);
  normpix_kernel<<<dim3(  64 / 8, BATCH), 256, 0, stream>>>(raw3,  f2n3,   64);

  // 4) WMMA correlation
  corr_kernel<<<dim3(NQ / 16, BATCH), 128, 0, stream>>>(
      coords2, feat1h, f2n0, f2n1, f2n2, f2n3, out);
}
